// GramBlock_18614388261663
// MI455X (gfx1250) — matrix operand order repaired and checked by reading, not compiled, not run on hardware
//
#include <hip/hip_runtime.h>
#include <hip/hip_bf16.h>

typedef __attribute__((ext_vector_type(16))) _Float16 v16h;
typedef __attribute__((ext_vector_type(8)))  float    v8f;

#define IH 1024
#define IW 1024
#define OH 1026
#define OW 1026
#define NPIX (OH * OW)
#define FT_STRIDE 34   // 32 + 2 halfs padding -> 68B row stride, conflict-free gathers

// ---------------------------------------------------------------------------
// Kernel 1: fused conv1 (3->32ch, 3x3, pad 2) + Gram accumulation via WMMA.
// Each wave: 32 pixels per iteration. Lane = pixel; computes all 32 channels
// of f for that pixel (fp32), stores f16 tile to LDS, gathers WMMA fragments,
// and accumulates the 32x32 Gram as three 16x16 f32 tiles (C10 = C01^T).
// Per-block partial Gram (32x32 f32) written deterministically to d_ws.
// ---------------------------------------------------------------------------
__global__ void __launch_bounds__(256)
conv_gram_kernel(const float* __restrict__ x,
                 const float* __restrict__ w1,
                 const float* __restrict__ b1,
                 float* __restrict__ partial)   // [gridDim.x][32*32]
{
    __shared__ float    sW[32 * 27];
    __shared__ float    sB[32];
    __shared__ _Float16 sF[8][32 * FT_STRIDE];   // per-wave f16 tile: [chan][pixel]
    __shared__ float    sG[8][3][16][16];        // per-wave de-swizzled Gram tiles

    const int tid  = threadIdx.x;   // 0..255
    const int wave = tid >> 5;      // 0..7
    const int lane = tid & 31;

    for (int i = tid; i < 32 * 27; i += 256) sW[i] = w1[i];
    if (tid < 32) sB[tid] = b1[tid];
    __syncthreads();

    v8f c00 = {}; v8f c01 = {}; v8f c11 = {};

    const int wavesTotal = (int)gridDim.x * 8;
    const int waveGlobal = (int)blockIdx.x * 8 + wave;
    const int nIter      = (NPIX + 32 * wavesTotal - 1) / (32 * wavesTotal);

    _Float16* ft = sF[wave];

    for (int it = 0; it < nIter; ++it) {
        const int p = (it * wavesTotal + waveGlobal) * 32 + lane;  // flat pixel

        float acc[32];
#pragma unroll
        for (int c = 0; c < 32; ++c) acc[c] = 0.f;

        if (p < NPIX) {
            const int oy = p / OW;
            const int ox = p % OW;
#pragma unroll
            for (int ic = 0; ic < 3; ++ic) {
#pragma unroll
                for (int ky = 0; ky < 3; ++ky) {
                    const int iy = oy - 2 + ky;
#pragma unroll
                    for (int kx = 0; kx < 3; ++kx) {
                        const int ix = ox - 2 + kx;
                        float xv = 0.f;
                        if (iy >= 0 && iy < IH && ix >= 0 && ix < IW)
                            xv = x[(ic * IH + iy) * IW + ix];
                        const int wofs = (ic * 3 + ky) * 3 + kx;
#pragma unroll
                        for (int c = 0; c < 32; ++c)
                            acc[c] = fmaf(xv, sW[c * 27 + wofs], acc[c]);
                    }
                }
            }
#pragma unroll
            for (int c = 0; c < 32; ++c) acc[c] += sB[c];
        }
        // Out-of-range lanes keep acc == 0 -> exact zero contribution to K.

#pragma unroll
        for (int c = 0; c < 32; ++c)
            ft[c * FT_STRIDE + lane] = (_Float16)acc[c];
        __syncthreads();

        // Gather WMMA fragments per CDNA5 ISA layouts.
        // A 16x32 f16 (M=chan row): lanes 0-15 -> K {0..7,16..23}; lanes 16-31 -> K {8..15,24..31}
        // B 32x16 f16 (N=chan col): lanes 0-15 -> K {0..7,16..23}; lanes 16-31 -> K {8..15,24..31}
        v16h a_lo, a_hi, b_lo, b_hi;
        const int m      = lane & 15;
        const int kaBase = (lane < 16) ? 0 : 8;
        const int kbBase = (lane < 16) ? 0 : 8;
#pragma unroll
        for (int j = 0; j < 16; ++j) {
            const int ka = kaBase + ((j < 8) ? j : (j + 8));
            a_lo[j] = ft[m * FT_STRIDE + ka];
            a_hi[j] = ft[(m + 16) * FT_STRIDE + ka];
            const int kb = kbBase + ((j < 8) ? j : (j + 8));
            b_lo[j] = ft[m * FT_STRIDE + kb];
            b_hi[j] = ft[(m + 16) * FT_STRIDE + kb];
        }
        __syncthreads();

        // Gram accumulation: D = A*B + C, f32 accumulators.
        c00 = __builtin_amdgcn_wmma_f32_16x16x32_f16(false, a_lo, false, b_lo,
                                                     (short)0, c00, false, false);
        c01 = __builtin_amdgcn_wmma_f32_16x16x32_f16(false, a_lo, false, b_hi,
                                                     (short)0, c01, false, false);
        c11 = __builtin_amdgcn_wmma_f32_16x16x32_f16(false, a_hi, false, b_hi,
                                                     (short)0, c11, false, false);
    }

    // De-swizzle C fragments: VGPR v, lane t -> M = v + (t<16 ? 0 : 8), N = t&15
    {
        const int n     = lane & 15;
        const int rbase = (lane < 16) ? 0 : 8;
#pragma unroll
        for (int v = 0; v < 8; ++v) {
            const int mr = rbase + v;
            sG[wave][0][mr][n] = c00[v];
            sG[wave][1][mr][n] = c01[v];
            sG[wave][2][mr][n] = c11[v];
        }
    }
    __syncthreads();

    // Reduce 8 waves -> full 32x32 block-partial Gram (use symmetry for C10).
    float* gp = partial + (size_t)blockIdx.x * 1024;
    for (int i = tid; i < 3 * 256; i += 256) {
        const int t  = i >> 8;
        const int mr = (i >> 4) & 15;
        const int nc = i & 15;
        float s = 0.f;
#pragma unroll
        for (int w = 0; w < 8; ++w) s += sG[w][t][mr][nc];
        if (t == 0)      gp[mr * 32 + nc] = s;
        else if (t == 2) gp[(16 + mr) * 32 + (16 + nc)] = s;
        else { gp[mr * 32 + (16 + nc)] = s; gp[(16 + nc) * 32 + mr] = s; }
    }
}

// ---------------------------------------------------------------------------
// Kernel 2: deterministic partial reduction + the tiny tail:
// conv2(1->16, s2, p2) + BN + relu ; conv3(16->32, s2, p2) + BN + relu ; mean.
// One block, everything in LDS.
// ---------------------------------------------------------------------------
__global__ void __launch_bounds__(256)
tail_kernel(const float* __restrict__ partial, int nParts,
            const float* __restrict__ w2, const float* __restrict__ b2,
            const float* __restrict__ g2, const float* __restrict__ bt2,
            const float* __restrict__ w3, const float* __restrict__ b3,
            const float* __restrict__ g3, const float* __restrict__ bt3,
            float* __restrict__ out)
{
    __shared__ float gram[32][32];
    __shared__ float h1[16][17][17];
    __shared__ float h2[32][10][10];
    __shared__ float mu[16], iv[16];

    const int tid = threadIdx.x;

    // Fixed-order reduction of block partials (bitwise deterministic).
    for (int i = tid; i < 1024; i += 256) {
        float s = 0.f;
        for (int p = 0; p < nParts; ++p) s += partial[(size_t)p * 1024 + i];
        gram[i >> 5][i & 31] = s;
    }
    __syncthreads();

    // conv2: 1ch 32x32 -> 16ch 17x17, stride 2, pad 2, k3
    for (int i = tid; i < 16 * 17 * 17; i += 256) {
        const int ch = i / 289, r = i % 289, oy = r / 17, ox = r % 17;
        float a = b2[ch];
        for (int ky = 0; ky < 3; ++ky) {
            const int iy = oy * 2 - 2 + ky;
            if (iy < 0 || iy >= 32) continue;
            for (int kx = 0; kx < 3; ++kx) {
                const int ix = ox * 2 - 2 + kx;
                if (ix < 0 || ix >= 32) continue;
                a += gram[iy][ix] * w2[(ch * 3 + ky) * 3 + kx];
            }
        }
        h1[ch][oy][ox] = a;
    }
    __syncthreads();

    // BN stats for conv2 output (per channel over 289 elems, ddof=0).
    if (tid < 16) {
        float m = 0.f, v = 0.f;
        for (int j = 0; j < 289; ++j) {
            const float t = h1[tid][j / 17][j % 17];
            m += t; v += t * t;
        }
        m /= 289.f; v = v / 289.f - m * m;
        mu[tid] = m; iv[tid] = rsqrtf(v + 1e-5f);
    }
    __syncthreads();

    // normalize + scale/shift + relu (in place)
    for (int i = tid; i < 16 * 289; i += 256) {
        const int ch = i / 289, r = i % 289;
        float t = (h1[ch][r / 17][r % 17] - mu[ch]) * iv[ch];
        t = g2[ch] * t + bt2[ch];
        h1[ch][r / 17][r % 17] = t > 0.f ? t : 0.f;
    }
    __syncthreads();

    // conv3: 16ch 17x17 -> 32ch 10x10, stride 2, pad 2, k3
    for (int i = tid; i < 32 * 100; i += 256) {
        const int ch = i / 100, r = i % 100, oy = r / 10, ox = r % 10;
        float a = b3[ch];
        for (int ic = 0; ic < 16; ++ic)
            for (int ky = 0; ky < 3; ++ky) {
                const int iy = oy * 2 - 2 + ky;
                if (iy < 0 || iy >= 17) continue;
                for (int kx = 0; kx < 3; ++kx) {
                    const int ix = ox * 2 - 2 + kx;
                    if (ix < 0 || ix >= 17) continue;
                    a += h1[ic][iy][ix] * w3[((ch * 16 + ic) * 3 + ky) * 3 + kx];
                }
            }
        h2[ch][oy][ox] = a;
    }
    __syncthreads();

    // BN + relu + spatial mean -> 32 outputs
    if (tid < 32) {
        float m = 0.f, v = 0.f;
        for (int j = 0; j < 100; ++j) {
            const float t = h2[tid][j / 10][j % 10];
            m += t; v += t * t;
        }
        m /= 100.f; v = v / 100.f - m * m;
        const float inv = rsqrtf(v + 1e-5f);
        float s = 0.f;
        for (int j = 0; j < 100; ++j) {
            float t = (h2[tid][j / 10][j % 10] - m) * inv;
            t = g3[tid] * t + bt3[tid];
            s += (t > 0.f ? t : 0.f);
        }
        out[tid] = s * 0.01f;
    }
}

extern "C" void kernel_launch(void* const* d_in, const int* in_sizes, int n_in,
                              void* d_out, int out_size, void* d_ws, size_t ws_size,
                              hipStream_t stream) {
    const float* x   = (const float*)d_in[0];
    const float* w1  = (const float*)d_in[1];
    const float* b1  = (const float*)d_in[2];
    const float* w2  = (const float*)d_in[3];
    const float* b2  = (const float*)d_in[4];
    const float* g2  = (const float*)d_in[5];
    const float* bt2 = (const float*)d_in[6];
    const float* w3  = (const float*)d_in[7];
    const float* b3  = (const float*)d_in[8];
    const float* g3  = (const float*)d_in[9];
    const float* bt3 = (const float*)d_in[10];
    float* out = (float*)d_out;
    float* partial = (float*)d_ws;

    // One 32x32 f32 partial Gram per block; clamp block count to scratch size.
    int nblocks = (int)(ws_size / (1024 * sizeof(float)));
    if (nblocks > 256) nblocks = 256;
    if (nblocks < 1)   nblocks = 1;

    conv_gram_kernel<<<nblocks, 256, 0, stream>>>(x, w1, b1, partial);
    tail_kernel<<<1, 256, 0, stream>>>(partial, nblocks,
                                       w2, b2, g2, bt2, w3, b3, g3, bt3, out);
}
